// AttentionClassifier_18657337934541
// MI455X (gfx1250) — matrix operand order repaired and checked by reading, not compiled, not run on hardware
//
#include <hip/hip_runtime.h>
#include <hip/hip_bf16.h>

// ---------------------------------------------------------------------------
// AttentionClassifier for MI455X (gfx1250, wave32, WMMA + Tensor Data Mover)
//   B=32, S=1024, D=768, V=32000
//   outputs: logits (32x2 f32) || a (32x1024x1024 f32), concatenated flat
// ---------------------------------------------------------------------------

typedef __attribute__((ext_vector_type(16))) _Float16 v16h;
typedef __attribute__((ext_vector_type(8)))  _Float16 v8h;
typedef __attribute__((ext_vector_type(8)))  float    v8f;
typedef __attribute__((ext_vector_type(4)))  unsigned u32x4;
typedef __attribute__((ext_vector_type(8)))  int      i32x8;
typedef __attribute__((ext_vector_type(4)))  int      i32x4;

#define BSZ  32
#define SSZ  1024
#define DM   768
#define NROW (BSZ * SSZ)   // 32768
#define KSTEP 32
#define NK   (DM / KSTEP)  // 24

// ---------------------------------------------------------------------------
// Tensor Data Mover: issue a 2-D f16 tile load (tile_h rows x tile_w elems,
// global row stride row_stride elems) into LDS at byte offset lds_addr.
// D# packing per CDNA5 ISA (group0) / §8.4 (group1):
//   g0: count=1 | lds_addr[63:32] | global_addr[120:64] | type=2[127:126]
//   g1: wg_mask=0 | data_size=1(2B) | tensor_dim0/1 | tile_dim0/1 | dim0_stride
// Groups 2/3 zero-filled (2-D tile: tile_dim2/3 unused).
// ---------------------------------------------------------------------------
__device__ __forceinline__ unsigned lds_off(const void* p) {
  // flat shared address: LDS aperture lives in addr[63:32]; addr[31:0] is the
  // wave-relative LDS byte offset (ISA 10.2 aperture table).
  return (unsigned)(unsigned long long)p;
}

__device__ __forceinline__ void tdm_load_2d(unsigned lds_addr, const void* gptr,
                                            unsigned tile_w, unsigned tile_h,
                                            unsigned tensor_w, unsigned tensor_h,
                                            unsigned row_stride) {
  const unsigned long long ga = (unsigned long long)gptr;
  u32x4 g0;
  g0[0] = 1u;                                               // count=1, user D#
  g0[1] = lds_addr;                                         // lds_addr (bytes)
  g0[2] = (unsigned)(ga & 0xFFFFFFFFull);                   // global_addr[31:0]
  g0[3] = (unsigned)((ga >> 32) & 0x1FFFFFFull) | (2u << 30); // ga[56:32]|type=2
  i32x8 g1;
  g1[0] = (int)(1u << 16);                                  // data_size=1 (2B)
  g1[1] = (int)((tensor_w & 0xFFFFu) << 16);                // tensor_dim0[15:0]
  g1[2] = (int)(((tensor_w >> 16) & 0xFFFFu) | ((tensor_h & 0xFFFFu) << 16));
  g1[3] = (int)(((tensor_h >> 16) & 0xFFFFu) | ((tile_w & 0xFFFFu) << 16));
  g1[4] = (int)(tile_h & 0xFFFFu);                          // tile_dim1, dim2=0
  g1[5] = (int)row_stride;                                  // dim0_stride[31:0]
  g1[6] = 0;                                                // stride hi / dim1_stride
  g1[7] = 0;
  const i32x4 z4 = {0, 0, 0, 0};
#if defined(__clang_major__) && (__clang_major__ >= 23)
  const i32x8 z8 = {0, 0, 0, 0, 0, 0, 0, 0};
  __builtin_amdgcn_tensor_load_to_lds(g0, g1, z4, z4, z8, 0);
#else
  __builtin_amdgcn_tensor_load_to_lds(g0, g1, z4, z4, 0);
#endif
}

// Build the 16x32 f16 A-fragment for one lane from an LDS row of 32 halves.
// Lanes 0-15 hold K=0..15, lanes 16-31 hold K=16..31 (contiguous, matches B);
// M = lane%16.
__device__ __forceinline__ v16h make_afrag(const _Float16* row, int half) {
  union { v16h v; v8h h[2]; } u;
  u.h[0] = *(const v8h*)(row + half * 16);
  u.h[1] = *(const v8h*)(row + half * 16 + 8);
  return u.v;
}

// ---------------------------------------------------------------------------
// x = (f16) emb[tokens]   (gather + downconvert)
// ---------------------------------------------------------------------------
__global__ __launch_bounds__(256)
void k_gather(const int* __restrict__ tokens, const float* __restrict__ emb,
              _Float16* __restrict__ Xh) {
  const int m = blockIdx.x, tid = threadIdx.x;
  const int tok = tokens[m];
  const float* e = emb + (size_t)tok * DM;
  _Float16* xr = Xh + (size_t)m * DM;
  xr[tid]       = (_Float16)e[tid];
  xr[tid + 256] = (_Float16)e[tid + 256];
  xr[tid + 512] = (_Float16)e[tid + 512];
}

// ---------------------------------------------------------------------------
// WT[n][k] = (f16) W[k][n] : tiled 32x32 transpose so GEMM B-tiles are
// row-major [n][k] and TDM can stream them (removes scalar LDS transposes).
// ---------------------------------------------------------------------------
__global__ __launch_bounds__(256)
void k_cvt_t(const float* __restrict__ src, _Float16* __restrict__ dst) {
  __shared__ float t[32][33];
  const int bx = blockIdx.x * 32, by = blockIdx.y * 32;
  const int tx = threadIdx.x & 31, ty = threadIdx.x >> 5;  // ty 0..7
  #pragma unroll
  for (int r = 0; r < 32; r += 8)
    t[ty + r][tx] = src[(size_t)(by + ty + r) * DM + bx + tx];
  __syncthreads();
  #pragma unroll
  for (int r = 0; r < 32; r += 8)
    dst[(size_t)(bx + ty + r) * DM + by + tx] = (_Float16)t[tx][ty + r];
}

// ---------------------------------------------------------------------------
// C_f16(NROW x 768) = X_f16(NROW x 768) @ W (given as WT[n][k], f16)
// 256 threads = 8 waves; tile 128(M) x 64(N); K-step 32.
// TDM double-buffered staging: wave 0 issues next tile pair, waits
// s_wait_tensorcnt<=2 (in-order) so current tile is resident, all waves WMMA.
// ---------------------------------------------------------------------------
__global__ __launch_bounds__(256)
void k_gemm_xw(const _Float16* __restrict__ X, const _Float16* __restrict__ WT,
               _Float16* __restrict__ Out) {
  __shared__ __attribute__((aligned(32))) _Float16 ldsX[2][128 * 32];  // [m][k]
  __shared__ __attribute__((aligned(32))) _Float16 ldsW[2][64 * 32];   // [n][k]
  const int tid = threadIdx.x;
  const int lane = tid & 31, wave = tid >> 5, half = lane >> 4;
  const int m0 = blockIdx.x * 128;
  const int n0 = blockIdx.y * 64;
  const _Float16* Xt = X + (size_t)m0 * DM;   // tile row start
  const _Float16* Wt = WT + (size_t)n0 * DM;

  v8f acc[4] = {};

  if (tid < 32) {  // prologue: stage K-step 0 into buffer 0
    tdm_load_2d(lds_off(&ldsX[0][0]), Xt, KSTEP, 128, DM, NROW, DM);
    tdm_load_2d(lds_off(&ldsW[0][0]), Wt, KSTEP, 64, DM, DM, DM);
  }

  for (int i = 0; i < NK; ++i) {
    const int cur = i & 1;
    if (tid < 32) {
      if (i + 1 < NK) {  // issue next tile pair, then wait for current pair
        const int kb = (i + 1) * KSTEP;
        tdm_load_2d(lds_off(&ldsX[cur ^ 1][0]), Xt + kb, KSTEP, 128, DM, NROW, DM);
        tdm_load_2d(lds_off(&ldsW[cur ^ 1][0]), Wt + kb, KSTEP, 64, DM, DM, DM);
        __builtin_amdgcn_s_wait_tensorcnt(2);
      } else {
        __builtin_amdgcn_s_wait_tensorcnt(0);
      }
    }
    __syncthreads();   // current tile visible to all waves

    const v16h a = make_afrag(&ldsX[cur][(wave * 16 + (lane & 15)) * 32], half);
    #pragma unroll
    for (int c = 0; c < 4; ++c) {
      // B: N = lane%16, K = (lane/16)*16 + j  -> contiguous 16 halves
      const v16h b = *(const v16h*)&ldsW[cur][(c * 16 + (lane & 15)) * 32 + half * 16];
      acc[c] = __builtin_amdgcn_wmma_f32_16x16x32_f16(
          false, a, false, b, (short)0, acc[c], false, false);
    }
    __syncthreads();   // all reads done before this buffer is DMA-overwritten
  }

  #pragma unroll
  for (int c = 0; c < 4; ++c) {
    const int n = n0 + c * 16 + (lane & 15);
    #pragma unroll
    for (int r = 0; r < 8; ++r) {
      const int m = m0 + wave * 16 + half * 8 + r;  // C layout: M=r+8*(lane/16)
      Out[(size_t)m * DM + n] = (_Float16)acc[c][r];
    }
  }
}

// ---------------------------------------------------------------------------
// s[b,i,j] = sum_d q[b,i,d]*k[b,j,d] - 1e9*mask[b,j]  (into `a` output region)
// K is already row-major in d -> both tiles TDM-streamed with no transpose.
// ---------------------------------------------------------------------------
__global__ __launch_bounds__(256)
void k_attn_scores(const _Float16* __restrict__ Q, const _Float16* __restrict__ K,
                   const int* __restrict__ mask, float* __restrict__ A) {
  __shared__ __attribute__((aligned(32))) _Float16 ldsQ[2][128 * 32];  // [i][d]
  __shared__ __attribute__((aligned(32))) _Float16 ldsK[2][64 * 32];   // [j][d]
  __shared__ float ldsMask[64];
  const int tid = threadIdx.x;
  const int lane = tid & 31, wave = tid >> 5, half = lane >> 4;
  const int b = blockIdx.z;
  const int i0 = blockIdx.x * 128;
  const int j0 = blockIdx.y * 64;
  const _Float16* Qt = Q + (size_t)(b * SSZ + i0) * DM;
  const _Float16* Kt = K + (size_t)(b * SSZ + j0) * DM;

  if (tid < 64) ldsMask[tid] = -1.0e9f * (float)mask[b * SSZ + j0 + tid];

  v8f acc[4] = {};

  if (tid < 32) {
    tdm_load_2d(lds_off(&ldsQ[0][0]), Qt, KSTEP, 128, DM, SSZ, DM);
    tdm_load_2d(lds_off(&ldsK[0][0]), Kt, KSTEP, 64, DM, SSZ, DM);
  }

  for (int i = 0; i < NK; ++i) {
    const int cur = i & 1;
    if (tid < 32) {
      if (i + 1 < NK) {
        const int kb = (i + 1) * KSTEP;
        tdm_load_2d(lds_off(&ldsQ[cur ^ 1][0]), Qt + kb, KSTEP, 128, DM, SSZ, DM);
        tdm_load_2d(lds_off(&ldsK[cur ^ 1][0]), Kt + kb, KSTEP, 64, DM, SSZ, DM);
        __builtin_amdgcn_s_wait_tensorcnt(2);
      } else {
        __builtin_amdgcn_s_wait_tensorcnt(0);
      }
    }
    __syncthreads();

    const v16h a = make_afrag(&ldsQ[cur][(wave * 16 + (lane & 15)) * 32], half);
    #pragma unroll
    for (int c = 0; c < 4; ++c) {
      const v16h bb = *(const v16h*)&ldsK[cur][(c * 16 + (lane & 15)) * 32 + half * 16];
      acc[c] = __builtin_amdgcn_wmma_f32_16x16x32_f16(
          false, a, false, bb, (short)0, acc[c], false, false);
    }
    __syncthreads();
  }

  float* Ab = A + (size_t)b * SSZ * SSZ;
  #pragma unroll
  for (int c = 0; c < 4; ++c) {
    const int j = j0 + c * 16 + (lane & 15);
    const float mk = ldsMask[c * 16 + (lane & 15)];
    #pragma unroll
    for (int r = 0; r < 8; ++r) {
      const int i = i0 + wave * 16 + half * 8 + r;
      Ab[(size_t)i * SSZ + j] = acc[c][r] + mk;
    }
  }
}

// ---------------------------------------------------------------------------
// In-place softmax over each row of 1024 (one workgroup per (b,i) row).
// ---------------------------------------------------------------------------
__global__ __launch_bounds__(256)
void k_softmax(float* __restrict__ A) {
  __shared__ float red[256];
  const int tid = threadIdx.x;
  float* row = A + (size_t)blockIdx.x * SSZ;
  float v0 = row[tid], v1 = row[tid + 256], v2 = row[tid + 512], v3 = row[tid + 768];
  float m = fmaxf(fmaxf(v0, v1), fmaxf(v2, v3));
  red[tid] = m; __syncthreads();
  for (int s = 128; s > 0; s >>= 1) {
    if (tid < s) red[tid] = fmaxf(red[tid], red[tid + s]);
    __syncthreads();
  }
  m = red[0]; __syncthreads();
  v0 = __expf(v0 - m); v1 = __expf(v1 - m);
  v2 = __expf(v2 - m); v3 = __expf(v3 - m);
  red[tid] = v0 + v1 + v2 + v3; __syncthreads();
  for (int s = 128; s > 0; s >>= 1) {
    if (tid < s) red[tid] += red[tid + s];
    __syncthreads();
  }
  const float inv = 1.0f / red[0];
  row[tid] = v0 * inv; row[tid + 256] = v1 * inv;
  row[tid + 512] = v2 * inv; row[tid + 768] = v3 * inv;
}

// ---------------------------------------------------------------------------
// cls[b,:] = sum_t a[b,0,t] * v[b,t,:]   (only row 0 of o is ever used)
// ---------------------------------------------------------------------------
__global__ __launch_bounds__(256)
void k_orow(const float* __restrict__ A, const _Float16* __restrict__ V,
            float* __restrict__ OCLS) {
  __shared__ float arow[SSZ];
  const int b = blockIdx.x, tid = threadIdx.x;
  const float* Ab = A + (size_t)b * SSZ * SSZ;  // row i=0
  for (int e = 0; e < 4; ++e) arow[tid + e * 256] = Ab[tid + e * 256];
  __syncthreads();
  float a0 = 0.f, a1 = 0.f, a2 = 0.f;
  const _Float16* Vb = V + (size_t)b * SSZ * DM;
  for (int t = 0; t < SSZ; ++t) {
    const float w = arow[t];
    const _Float16* vr = Vb + (size_t)t * DM;
    a0 += w * (float)vr[tid];
    a1 += w * (float)vr[tid + 256];
    a2 += w * (float)vr[tid + 512];
  }
  OCLS[b * DM + tid] = a0;
  OCLS[b * DM + tid + 256] = a1;
  OCLS[b * DM + tid + 512] = a2;
}

// ---------------------------------------------------------------------------
// LayerNorm -> h = gelu_exact(cls_n @ w1 + b1) -> logits = h @ wh + bh
// ---------------------------------------------------------------------------
__global__ __launch_bounds__(256)
void k_head(const float* __restrict__ OCLS, const float* __restrict__ ln_g,
            const float* __restrict__ ln_b, const float* __restrict__ w1,
            const float* __restrict__ b1, const float* __restrict__ wh,
            const float* __restrict__ bh, float* __restrict__ logits) {
  __shared__ float red[256];
  __shared__ float cn[DM];
  __shared__ float hh[DM];
  const int b = blockIdx.x, tid = threadIdx.x;
  const float x0 = OCLS[b * DM + tid];
  const float x1 = OCLS[b * DM + tid + 256];
  const float x2 = OCLS[b * DM + tid + 512];
  red[tid] = x0 + x1 + x2; __syncthreads();
  for (int s = 128; s > 0; s >>= 1) {
    if (tid < s) red[tid] += red[tid + s];
    __syncthreads();
  }
  const float mu = red[0] * (1.0f / DM); __syncthreads();
  const float d0 = x0 - mu, d1 = x1 - mu, d2 = x2 - mu;
  red[tid] = d0 * d0 + d1 * d1 + d2 * d2; __syncthreads();
  for (int s = 128; s > 0; s >>= 1) {
    if (tid < s) red[tid] += red[tid + s];
    __syncthreads();
  }
  const float rs = rsqrtf(red[0] * (1.0f / DM) + 1e-5f); __syncthreads();
  cn[tid]       = d0 * rs * ln_g[tid]       + ln_b[tid];
  cn[tid + 256] = d1 * rs * ln_g[tid + 256] + ln_b[tid + 256];
  cn[tid + 512] = d2 * rs * ln_g[tid + 512] + ln_b[tid + 512];
  __syncthreads();
  for (int e = 0; e < 3; ++e) {
    const int n = tid + e * 256;
    float accv = b1[n];
    for (int k = 0; k < DM; ++k) accv += cn[k] * w1[(size_t)k * DM + n];
    hh[n] = 0.5f * accv * (1.0f + erff(accv * 0.70710678118654752f));  // exact gelu
  }
  __syncthreads();
  float p0 = 0.f, p1 = 0.f;
  for (int e = 0; e < 3; ++e) {
    const int n = tid + e * 256;
    p0 += hh[n] * wh[n * 2];
    p1 += hh[n] * wh[n * 2 + 1];
  }
  red[tid] = p0; __syncthreads();
  for (int s = 128; s > 0; s >>= 1) {
    if (tid < s) red[tid] += red[tid + s];
    __syncthreads();
  }
  const float l0 = red[0]; __syncthreads();
  red[tid] = p1; __syncthreads();
  for (int s = 128; s > 0; s >>= 1) {
    if (tid < s) red[tid] += red[tid + s];
    __syncthreads();
  }
  if (tid == 0) {
    logits[b * 2]     = l0 + bh[0];
    logits[b * 2 + 1] = red[0] + bh[1];
  }
}

// ---------------------------------------------------------------------------
extern "C" void kernel_launch(void* const* d_in, const int* in_sizes, int n_in,
                              void* d_out, int out_size, void* d_ws, size_t ws_size,
                              hipStream_t stream) {
  (void)in_sizes; (void)n_in; (void)out_size; (void)ws_size;
  const int*   tokens = (const int*)d_in[0];
  const int*   mask   = (const int*)d_in[1];
  const float* emb    = (const float*)d_in[2];
  const float* wq     = (const float*)d_in[3];
  const float* wk     = (const float*)d_in[4];
  const float* wv     = (const float*)d_in[5];
  const float* ln_g   = (const float*)d_in[6];
  const float* ln_b   = (const float*)d_in[7];
  const float* w1     = (const float*)d_in[8];
  const float* b1     = (const float*)d_in[9];
  const float* wh     = (const float*)d_in[10];
  const float* bh     = (const float*)d_in[11];

  float* logits = (float*)d_out;
  float* Aout   = logits + BSZ * 2;   // 32*1024*1024 f32 attention output

  // workspace: f16 intermediates then f32 cls rows (~205 MB)
  _Float16* Xh  = (_Float16*)d_ws;
  _Float16* Qh  = Xh + (size_t)NROW * DM;
  _Float16* Kh  = Qh + (size_t)NROW * DM;
  _Float16* Vh  = Kh + (size_t)NROW * DM;
  _Float16* WqT = Vh + (size_t)NROW * DM;   // transposed f16 weights [n][k]
  _Float16* WkT = WqT + (size_t)DM * DM;
  _Float16* WvT = WkT + (size_t)DM * DM;
  float*    OCLS = (float*)(WvT + (size_t)DM * DM);

  k_gather<<<dim3(NROW), dim3(256), 0, stream>>>(tokens, emb, Xh);
  k_cvt_t<<<dim3(DM / 32, DM / 32), dim3(256), 0, stream>>>(wq, WqT);
  k_cvt_t<<<dim3(DM / 32, DM / 32), dim3(256), 0, stream>>>(wk, WkT);
  k_cvt_t<<<dim3(DM / 32, DM / 32), dim3(256), 0, stream>>>(wv, WvT);

  k_gemm_xw<<<dim3(NROW / 128, DM / 64), dim3(256), 0, stream>>>(Xh, WqT, Qh);
  k_gemm_xw<<<dim3(NROW / 128, DM / 64), dim3(256), 0, stream>>>(Xh, WkT, Kh);
  k_gemm_xw<<<dim3(NROW / 128, DM / 64), dim3(256), 0, stream>>>(Xh, WvT, Vh);

  k_attn_scores<<<dim3(SSZ / 128, SSZ / 64, BSZ), dim3(256), 0, stream>>>(Qh, Kh, mask, Aout);
  k_softmax<<<dim3(BSZ * SSZ), dim3(256), 0, stream>>>(Aout);
  k_orow<<<dim3(BSZ), dim3(256), 0, stream>>>(Aout, Vh, OCLS);
  k_head<<<dim3(BSZ), dim3(256), 0, stream>>>(OCLS, ln_g, ln_b, w1, b1, wh, bh, logits);
}
